// BitLinear_20315195310829
// MI455X (gfx1250) — 8-bit matrix operand order repaired and checked by reading, compiled, not run on hardware
//
#include <hip/hip_runtime.h>
#include <math.h>


#define IN_F   2048
#define OUT_F  8192
#define MROWS  8192
#define NW     16777216u
#define LSCALE 0.022097086912079608f
#define EPSF   1e-8f

typedef __attribute__((ext_vector_type(16))) int   v16i;
typedef __attribute__((ext_vector_type(8)))  float v8f;

#if defined(__HIP_DEVICE_COMPILE__) && defined(__has_builtin)
#if __has_builtin(__builtin_amdgcn_cvt_pk_fp8_f32)
#define HAVE_CVT_FP8 1
#endif
#if __has_builtin(__builtin_amdgcn_global_load_async_to_lds_b128) && \
    __has_builtin(__builtin_amdgcn_global_load_async_to_lds_b64)
#define HAVE_ASYNC_LDS 1
#endif
#endif

#ifdef HAVE_ASYNC_LDS
typedef int v2i_t __attribute__((vector_size(8)));
typedef int v4i_t __attribute__((vector_size(16)));
typedef __attribute__((address_space(1))) v2i_t* g64p;
typedef __attribute__((address_space(3))) v2i_t* l64p;
typedef __attribute__((address_space(1))) v4i_t* g128p;
typedef __attribute__((address_space(3))) v4i_t* l128p;

__device__ inline l64p  lds64(void* p)  { return (l64p)(unsigned int)(uintptr_t)p; }
__device__ inline l128p lds128(void* p) { return (l128p)(unsigned int)(uintptr_t)p; }
__device__ inline g64p  glb64(const void* p)  { return (g64p)(uintptr_t)p; }
__device__ inline g128p glb128(const void* p) { return (g128p)(uintptr_t)p; }

__device__ inline void wait_async0() {
#if __has_builtin(__builtin_amdgcn_s_wait_asynccnt)
  __builtin_amdgcn_s_wait_asynccnt(0);
#else
  asm volatile("s_wait_asynccnt 0x0" ::: "memory");
#endif
}
#endif

__device__ inline unsigned f32_to_e4m3(float x) {
  unsigned u = __float_as_uint(x);
  unsigned s = (u >> 24) & 0x80u;
  float a = fabsf(x);
  if (a >= 448.0f) return s | 0x7Eu;
  if (a < 0.015625f) {
    int dq = (int)rintf(a * 512.0f);
    return s | (unsigned)dq;
  }
  int e; float mf = frexpf(a, &e);
  int q = (int)rintf(mf * 16.0f);
  if (q == 16) { q = 8; e += 1; }
  int E = (e - 1) + 7;
  if (E >= 16) return s | 0x7Eu;
  return s | ((unsigned)E << 3) | (unsigned)(q - 8);
}

__global__ void zero1_kernel(float* p) { p[0] = 0.0f; }

__global__ __launch_bounds__(256) void absum_kernel(const float* __restrict__ W,
                                                    float* __restrict__ sum) {
  __shared__ float red[256];
  int t = threadIdx.x;
  unsigned idx = blockIdx.x * 256u + t;
  unsigned stride = gridDim.x * 256u;
  float s = 0.0f;
  for (unsigned i = idx; i < NW; i += stride) s += fabsf(W[i]);
  red[t] = s; __syncthreads();
  for (int o = 128; o > 0; o >>= 1) {
    if (t < o) red[t] += red[t + o];
    __syncthreads();
  }
  if (t == 0) atomicAdd(sum, red[0]);
}

__global__ __launch_bounds__(256) void quantx_kernel(const float* __restrict__ x,
                                                     unsigned char* __restrict__ xq,
                                                     float* __restrict__ a_scale) {
  __shared__ float red[256];
  const int row = blockIdx.x;
  const int t = threadIdx.x;
  const float* xr = x + (size_t)row * IN_F;
  float v[8]; float am = 0.0f;
#pragma unroll
  for (int i = 0; i < 8; ++i) { v[i] = xr[t * 8 + i]; am = fmaxf(am, fabsf(v[i])); }
  red[t] = am; __syncthreads();
  for (int o = 128; o > 0; o >>= 1) {
    if (t < o) red[t] = fmaxf(red[t], red[t + o]);
    __syncthreads();
  }
  float amax = fmaxf(red[0], 1e-20f);
  if (t == 0) a_scale[row] = amax * (1.0f / 448.0f);
  float inv = 448.0f / amax;
  unsigned p0, p1;
#ifdef HAVE_CVT_FP8
  int w0 = __builtin_amdgcn_cvt_pk_fp8_f32(v[0] * inv, v[1] * inv, 0, false);
  w0     = __builtin_amdgcn_cvt_pk_fp8_f32(v[2] * inv, v[3] * inv, w0, true);
  int w1 = __builtin_amdgcn_cvt_pk_fp8_f32(v[4] * inv, v[5] * inv, 0, false);
  w1     = __builtin_amdgcn_cvt_pk_fp8_f32(v[6] * inv, v[7] * inv, w1, true);
  p0 = (unsigned)w0; p1 = (unsigned)w1;
#else
  p0 = f32_to_e4m3(v[0] * inv) | (f32_to_e4m3(v[1] * inv) << 8) |
       (f32_to_e4m3(v[2] * inv) << 16) | (f32_to_e4m3(v[3] * inv) << 24);
  p1 = f32_to_e4m3(v[4] * inv) | (f32_to_e4m3(v[5] * inv) << 8) |
       (f32_to_e4m3(v[6] * inv) << 16) | (f32_to_e4m3(v[7] * inv) << 24);
#endif
  ((uint2*)(xq + (size_t)row * IN_F))[t] = make_uint2(p0, p1);
}

__device__ inline unsigned tern_code(float f, float inv_beta) {
  float r = rintf(f * inv_beta);
  return (r >= 0.5f) ? 0x38u : ((r <= -0.5f) ? 0xB8u : 0x00u);
}

__global__ __launch_bounds__(256) void quantw_kernel(const float* __restrict__ W,
                                                     const float* __restrict__ beta_sum,
                                                     unsigned char* __restrict__ wq) {
  const float beta = fmaxf(beta_sum[0] * (1.0f / 16777216.0f), EPSF);
  const float inv = 1.0f / beta;
  unsigned i4 = blockIdx.x * 256u + threadIdx.x;
  float4 w = ((const float4*)W)[i4];
  unsigned c = tern_code(w.x, inv) | (tern_code(w.y, inv) << 8) |
               (tern_code(w.z, inv) << 16) | (tern_code(w.w, inv) << 24);
  ((unsigned*)wq)[i4] = c;
}

__global__ __launch_bounds__(256) void bitlinear_gemm(
    const unsigned char* __restrict__ xq,
    const unsigned char* __restrict__ wq,
    const float* __restrict__ a_scale,
    const float* __restrict__ bias,
    const float* __restrict__ beta_sum,
    float* __restrict__ out) {
  __shared__ __align__(64) unsigned char sAs[2][256][64];
  __shared__ __align__(64) unsigned char sBs[2][128][64];
  __shared__ float sScale[128];
  __shared__ float sBias[64];

  const int t = threadIdx.x;
  const int lane = t & 31;
  const int wave = t >> 5;
  const int waveM = wave & 3;
  const int waveN = wave >> 2;
  const int m0 = blockIdx.y * 128;
  const int n0 = blockIdx.x * 64;

  if (t < 128)      sScale[t] = a_scale[m0 + t];
  else if (t < 192) sBias[t - 128] = bias[n0 + (t - 128)];

  const int lrow = t >> 3;
  const int lcol = t & 7;
  const unsigned char* gA = xq + (size_t)(m0 + lrow) * IN_F + lcol * 16;
  const unsigned char* gB = wq + (size_t)(n0 + lrow) * IN_F + lcol * 16;

  const int slotA0 = (lrow >> 4) * 32 + (lrow & 15) + 16 * (lcol & 1);
  const int dA     = (lcol >> 1) * 4;
  const int slotB0 = (lrow >> 4) * 32 + (lrow & 15) + 16 * (lcol & 1);
  const int dB     = (lcol >> 1) * 4;

#ifdef HAVE_ASYNC_LDS
  auto stage = [&](int buf, int k) {
#pragma unroll
    for (int p = 0; p < 4; ++p) {
      const unsigned char* g = gA + (size_t)p * 32 * IN_F + k;
      __builtin_amdgcn_global_load_async_to_lds_b64(
          glb64(g), lds64(&sAs[buf][slotA0 + 64 * p][dA * 4]), 0, 0);
      __builtin_amdgcn_global_load_async_to_lds_b64(
          glb64(g + 8), lds64(&sAs[buf][slotA0 + 64 * p][dA * 4 + 8]), 0, 0);
    }
#pragma unroll
    for (int p = 0; p < 2; ++p) {
      const unsigned char* g = gB + (size_t)p * 32 * IN_F + k;
      __builtin_amdgcn_global_load_async_to_lds_b128(
          glb128(g), lds128(&sBs[buf][slotB0 + 64 * p][dB * 4]), 0, 0);
    }
  };
#else
  uint4 ra[4], rb[2];
  auto gload = [&](int k) {
#pragma unroll
    for (int p = 0; p < 4; ++p) ra[p] = *(const uint4*)(gA + (size_t)p * 32 * IN_F + k);
#pragma unroll
    for (int p = 0; p < 2; ++p) rb[p] = *(const uint4*)(gB + (size_t)p * 32 * IN_F + k);
  };
  auto lstore = [&](int buf) {
#pragma unroll
    for (int p = 0; p < 4; ++p) {
      *(uint2*)&sAs[buf][slotA0 + 64 * p][dA * 4]      = make_uint2(ra[p].x, ra[p].y);
      *(uint2*)&sAs[buf][slotA0 + 64 * p][dA * 4 + 8] = make_uint2(ra[p].z, ra[p].w);
    }
#pragma unroll
    for (int p = 0; p < 2; ++p) *(uint4*)&sBs[buf][slotB0 + 64 * p][dB * 4] = rb[p];
  };
#endif

#ifdef HAVE_ASYNC_LDS
  stage(0, 0);
  wait_async0();
#else
  gload(0);
  lstore(0);
#endif
  __syncthreads();

  v8f acc[2][2] = {};

  for (int kk = 0; kk < 16; ++kk) {
    const int cur = kk & 1;
#ifdef HAVE_ASYNC_LDS
    if (kk < 15) stage(cur ^ 1, (kk + 1) * 128);
#else
    if (kk < 15) {
      gload((kk + 1) * 128);
      if (kk < 14) {
        __builtin_prefetch(gA + (size_t)(kk + 2) * 128, 0, 3);
        __builtin_prefetch(gB + (size_t)(kk + 2) * 128, 0, 3);
      }
    }
#endif

    v16i bf0 = *(const v16i*)&sBs[cur][(waveN * 2 + 0) * 32 + lane][0];
    v16i bf1 = *(const v16i*)&sBs[cur][(waveN * 2 + 1) * 32 + lane][0];
#pragma unroll
    for (int f = 0; f < 2; ++f) {
      v16i af = *(const v16i*)&sAs[cur][(waveM * 2 + f) * 32 + lane][0];
      acc[f][0] = __builtin_amdgcn_wmma_f32_16x16x128_fp8_fp8(
          af, bf0, (short)0, acc[f][0], false, false);
      acc[f][1] = __builtin_amdgcn_wmma_f32_16x16x128_fp8_fp8(
          af, bf1, (short)0, acc[f][1], false, false);
    }

#ifdef HAVE_ASYNC_LDS
    if (kk < 15) wait_async0();
#else
    if (kk < 15) lstore(cur ^ 1);
#endif
    __syncthreads();
  }

  const float beta = fmaxf(beta_sum[0] * (1.0f / 16777216.0f), EPSF);
  const int half = lane >> 4;
#pragma unroll
  for (int f = 0; f < 2; ++f) {
#pragma unroll
    for (int g = 0; g < 2; ++g) {
      const int nl = waveN * 32 + g * 16 + (lane & 15);
      const float bv = sBias[nl];
#pragma unroll
      for (int r = 0; r < 8; ++r) {
        const int ml = waveM * 32 + f * 16 + half * 8 + r;
        out[(size_t)(m0 + ml) * OUT_F + (n0 + nl)] =
            (acc[f][g][r] * sScale[ml] * beta + bv) * LSCALE;
      }
    }
  }
}

__global__ __launch_bounds__(256) void naive_kernel(const float* __restrict__ x,
                                                    const float* __restrict__ W,
                                                    const float* __restrict__ b,
                                                    const float* __restrict__ beta_sum,
                                                    float* __restrict__ out) {
  const size_t idx = (size_t)blockIdx.x * 256 + threadIdx.x;
  const int m = (int)(idx >> 13);
  const int n = (int)(idx & 8191);
  const float beta = fmaxf(beta_sum[0] * (1.0f / 16777216.0f), EPSF);
  const float inv = 1.0f / beta;
  float acc = 0.0f;
  const float* xr = x + (size_t)m * IN_F;
  const float* wr = W + (size_t)n * IN_F;
  for (int k = 0; k < IN_F; ++k) {
    float r = rintf(wr[k] * inv);
    r = fminf(fmaxf(r, -1.0f), 1.0f);
    acc += xr[k] * r;
  }
  out[idx] = (acc * beta + b[n]) * LSCALE;
}

extern "C" void kernel_launch(void* const* d_in, const int* in_sizes, int n_in,
                              void* d_out, int out_size, void* d_ws, size_t ws_size,
                              hipStream_t stream) {
  const float* x = (const float*)d_in[0];
  const float* W = (const float*)d_in[1];
  const float* b = (const float*)d_in[2];
  float* out = (float*)d_out;

  float* beta_sum = (float*)d_ws;
  const size_t need = 256 + 32768 + 2ull * 16777216ull;

  if (ws_size < need) {
    if (ws_size < 4) return;
    zero1_kernel<<<1, 1, 0, stream>>>(beta_sum);
    absum_kernel<<<2048, 256, 0, stream>>>(W, beta_sum);
    naive_kernel<<<262144, 256, 0, stream>>>(x, W, b, beta_sum, out);
    return;
  }

  float* a_scale = (float*)((char*)d_ws + 256);
  unsigned char* xq = (unsigned char*)d_ws + 256 + 32768;
  unsigned char* wq = xq + 16777216ull;

  zero1_kernel<<<1, 1, 0, stream>>>(beta_sum);
  absum_kernel<<<2048, 256, 0, stream>>>(W, beta_sum);
  quantx_kernel<<<MROWS, 256, 0, stream>>>(x, xq, a_scale);
  quantw_kernel<<<16384, 256, 0, stream>>>(W, beta_sum, wq);

  dim3 grid(OUT_F / 64, MROWS / 128);
  bitlinear_gemm<<<grid, 256, 0, stream>>>(xq, wq, a_scale, b, beta_sum, out);
}
